// GCN_ID_38714835206185
// MI455X (gfx1250) — matrix operand order repaired and checked by reading, not compiled, not run on hardware
//
#include <hip/hip_runtime.h>
#include <hip/hip_bf16.h>

typedef __bf16 v8bf  __attribute__((ext_vector_type(8)));
typedef __bf16 v16bf __attribute__((ext_vector_type(16)));
typedef float  v8f   __attribute__((ext_vector_type(8)));

// ---------------------------------------------------------------- utilities

__global__ void zero_f32_kernel(float* __restrict__ p, long long n) {
    long long i = (long long)blockIdx.x * blockDim.x + threadIdx.x;
    long long stride = (long long)gridDim.x * blockDim.x;
    for (; i < n; i += stride) p[i] = 0.0f;
}

// degree count over dst (with self loop added later)
__global__ void deg_count_kernel(const long long* __restrict__ dst,
                                 float* __restrict__ deg, int E_) {
    int e = blockIdx.x * blockDim.x + threadIdx.x;
    if (e < E_) atomicAdd(&deg[(int)dst[e]], 1.0f);
}

// dis[n] = rsqrt(deg[n] + 1)  (in place)
__global__ void dis_kernel(float* __restrict__ deg, int N_) {
    int n = blockIdx.x * blockDim.x + threadIdx.x;
    if (n < N_) deg[n] = rsqrtf(deg[n] + 1.0f);
}

// convert W [K,NO] f32 row-major -> Wt [NO,K] bf16 (transposed, so WMMA B
// fragments become contiguous per-lane K-runs)
__global__ void w_to_bf16_t_kernel(const float* __restrict__ W,
                                   __bf16* __restrict__ Wt, int K, int NO) {
    int t = blockIdx.x * blockDim.x + threadIdx.x;
    if (t < K * NO) {
        int k = t / NO, no = t - k * NO;
        Wt[(size_t)no * K + k] = (__bf16)W[t];
    }
}

// ------------------------------------------------------- embedding frontend
// one block (128 threads) per node: x1=relu(x[:7]@W1+b1), x2=relu(x[7:]@W2+b2)
// hcat = [x1, x2, id_emb] in bf16
__global__ __launch_bounds__(128)
void emb_kernel(const float* __restrict__ x, const float* __restrict__ id_emb,
                const float* __restrict__ W1, const float* __restrict__ b1,
                const float* __restrict__ W2, const float* __restrict__ b2,
                __bf16* __restrict__ hcat, int N_) {
    int n = blockIdx.x;
    if (n >= N_) return;
    __shared__ float xs[19];
    int j = threadIdx.x;
    if (j < 19) xs[j] = x[(size_t)n * 19 + j];
    __syncthreads();
    float a1 = b1[j];
#pragma unroll
    for (int k = 0; k < 7; ++k) a1 += xs[k] * W1[k * 128 + j];
    float a2 = b2[j];
#pragma unroll
    for (int k = 0; k < 12; ++k) a2 += xs[7 + k] * W2[k * 128 + j];
    __bf16* hrow = hcat + (size_t)n * 320;
    hrow[j]       = (__bf16)fmaxf(a1, 0.0f);
    hrow[128 + j] = (__bf16)fmaxf(a2, 0.0f);
    if (j < 64) hrow[256 + j] = (__bf16)id_emb[(size_t)n * 64 + j];
}

// ------------------------------------------------------------- WMMA GEMM
// O[N,NO] = act(A[N,K](bf16) @ W[K,NO] + bias), W given transposed bf16 [NO,K].
// Block = 256 threads = 8 waves, block tile 64x32 (4x2 waves of 16x16).
template <bool OUT_BF16>
__global__ __launch_bounds__(256)
void wmma_gemm_kernel(const __bf16* __restrict__ A, const __bf16* __restrict__ Wt,
                      const float* __restrict__ bias, void* __restrict__ OutV,
                      int Nrows, int K, int NO, int do_relu) {
    const int wave = threadIdx.x >> 5;
    const int lane = threadIdx.x & 31;
    const int half = lane >> 4;   // lane half selects K sub-run
    const int mrow = lane & 15;   // A row / B column within tile

    const int row0 = blockIdx.x * 64 + (wave >> 1) * 16;
    const int col0 = blockIdx.y * 32 + (wave & 1) * 16;

    // clamp A row so loads stay in range; EXEC stays all-ones for WMMA
    int ar = row0 + mrow;
    if (ar > Nrows - 1) ar = Nrows - 1;
    const __bf16* aptr = A + (size_t)ar * K + half * 16;
    const __bf16* bptr = Wt + (size_t)(col0 + mrow) * K + half * 16;

    v8f acc = {};
    for (int k0 = 0; k0 < K; k0 += 32) {
        __builtin_prefetch(aptr + k0 + 64, 0, 3);  // global_prefetch_b8
        v8bf a_lo = *(const v8bf*)(aptr + k0);        // K = k0+half*16 .. +7
        v8bf a_hi = *(const v8bf*)(aptr + k0 + 8);    // K = k0+half*16+8 ..
        v8bf b_lo = *(const v8bf*)(bptr + k0);        // K = k0+half*16 .. +7
        v8bf b_hi = *(const v8bf*)(bptr + k0 + 8);    // K = k0+half*16+8 ..
        v16bf af = __builtin_shufflevector(a_lo, a_hi, 0, 1, 2, 3, 4, 5, 6, 7,
                                           8, 9, 10, 11, 12, 13, 14, 15);
        v16bf bfv = __builtin_shufflevector(b_lo, b_hi, 0, 1, 2, 3, 4, 5, 6, 7,
                                            8, 9, 10, 11, 12, 13, 14, 15);
        acc = __builtin_amdgcn_wmma_f32_16x16x32_bf16(
            false, af, false, bfv, (short)0, acc, false, false);
    }

    const int col = col0 + mrow;
    const float bv = bias ? bias[col] : 0.0f;
#pragma unroll
    for (int v = 0; v < 8; ++v) {
        int r = row0 + half * 8 + v;  // D layout: VGPR v -> row half*8+v
        if (r < Nrows) {
            float o = acc[v] + bv;
            if (do_relu) o = fmaxf(o, 0.0f);
            if (OUT_BF16)
                ((__bf16*)OutV)[(size_t)r * NO + col] = (__bf16)o;
            else
                ((float*)OutV)[(size_t)r * NO + col] = o;
        }
    }
}

// ---------------------------------------------------------- edge scatter
// one wave (32 lanes) per edge; lane handles 4 consecutive features
__global__ __launch_bounds__(256)
void scatter_kernel(const long long* __restrict__ src, const long long* __restrict__ dst,
                    const float* __restrict__ dis, const float* __restrict__ hw,
                    float* __restrict__ agg, int E_) {
    int t = blockIdx.x * blockDim.x + threadIdx.x;
    int e = t >> 5;
    int lane = t & 31;
    if (e >= E_) return;
    int s = (int)src[e], d = (int)dst[e];
    float norm = dis[s] * dis[d];
    const float4 v = *(const float4*)(hw + (size_t)s * 128 + lane * 4);
    float* ap = agg + (size_t)d * 128 + lane * 4;
    atomicAdd(ap + 0, v.x * norm);
    atomicAdd(ap + 1, v.y * norm);
    atomicAdd(ap + 2, v.z * norm);
    atomicAdd(ap + 3, v.w * norm);
}

// -------------------------------------- conv finalize + community pooling
// h = relu(agg + hw*dis^2 + b); pool_sum[community] += h; optional bf16 out
__global__ __launch_bounds__(128)
void finalize_kernel(const float* __restrict__ agg, const float* __restrict__ hw,
                     const float* __restrict__ dis, const float* __restrict__ bias,
                     const long long* __restrict__ community,
                     float* __restrict__ pool_sum, float* __restrict__ cnt,
                     __bf16* __restrict__ h_out, int N_, int count_flag) {
    int n = blockIdx.x;
    if (n >= N_) return;
    int j = threadIdx.x;
    float d = dis[n];
    float v = agg[(size_t)n * 128 + j] + hw[(size_t)n * 128 + j] * (d * d) + bias[j];
    v = fmaxf(v, 0.0f);
    int c = (int)community[n];
    atomicAdd(&pool_sum[(size_t)c * 128 + j], v);
    if (h_out) h_out[(size_t)n * 128 + j] = (__bf16)v;
    if (count_flag && j == 0) atomicAdd(&cnt[c], 1.0f);
}

// ------------------------------------------------------------------- head
// mean = pool_sum/cnt (= mean1+mean2); z=relu(mean@(Wl1[:128]+Wl1[128:])+b1)
// out[c] = z @ Wl2 + b2
__global__ __launch_bounds__(128)
void head_kernel(const float* __restrict__ pool_sum, const float* __restrict__ cnt,
                 const float* __restrict__ Wl1, const float* __restrict__ bl1,
                 const float* __restrict__ Wl2, const float* __restrict__ bl2,
                 float* __restrict__ out, int C_) {
    int c = blockIdx.x;
    if (c >= C_) return;
    int j = threadIdx.x;
    __shared__ float m[128];
    __shared__ float red[128];
    float ct = fmaxf(cnt[c], 1.0f);
    m[j] = pool_sum[(size_t)c * 128 + j] / ct;
    __syncthreads();
    float z = bl1[j];
    for (int k = 0; k < 128; ++k)
        z += m[k] * (Wl1[k * 128 + j] + Wl1[(k + 128) * 128 + j]);
    z = fmaxf(z, 0.0f);
    red[j] = z * Wl2[j];
    __syncthreads();
    for (int s = 64; s > 0; s >>= 1) {
        if (j < s) red[j] += red[j + s];
        __syncthreads();
    }
    if (j == 0) out[c] = red[0] + bl2[0];
}

// ---------------------------------------------------------------- launch

extern "C" void kernel_launch(void* const* d_in, const int* in_sizes, int n_in,
                              void* d_out, int out_size, void* d_ws, size_t ws_size,
                              hipStream_t stream) {
    (void)in_sizes; (void)n_in; (void)out_size; (void)ws_size;
    const int N = 100000, E = 1600000, C = 1000;

    const float*     x      = (const float*)d_in[0];
    const long long* ei     = (const long long*)d_in[1];
    const long long* src    = ei;
    const long long* dst    = ei + E;
    const long long* comm   = (const long long*)d_in[2];
    const float*     id_emb = (const float*)d_in[3];
    const float* W_emb1 = (const float*)d_in[4];  const float* b_emb1 = (const float*)d_in[5];
    const float* W_emb2 = (const float*)d_in[6];  const float* b_emb2 = (const float*)d_in[7];
    const float* W_emb3 = (const float*)d_in[8];  const float* b_emb3 = (const float*)d_in[9];
    const float* W_c1   = (const float*)d_in[10]; const float* b_c1   = (const float*)d_in[11];
    const float* W_c2   = (const float*)d_in[12]; const float* b_c2   = (const float*)d_in[13];
    const float* W_l1   = (const float*)d_in[14]; const float* b_l1   = (const float*)d_in[15];
    const float* W_l2   = (const float*)d_in[16]; const float* b_l2   = (const float*)d_in[17];
    float* out = (float*)d_out;

    // ---- workspace carve-up
    char* ws = (char*)d_ws;
    size_t off = 0;
    auto take = [&](size_t bytes) -> void* {
        off = (off + 255) & ~(size_t)255;
        void* p = ws + off;
        off += bytes;
        return p;
    };
    float*  dis  = (float*)take((size_t)N * 4);           // deg -> deg_inv_sqrt
    __bf16* hcat = (__bf16*)take((size_t)N * 320 * 2);    // [x1,x2,id] bf16
    __bf16* h320 = (__bf16*)take((size_t)N * 320 * 2);    // relu(emb3) bf16
    float*  hw   = (float*)take((size_t)N * 128 * 4);     // h@W_conv (reused)
    float*  agg  = (float*)take((size_t)N * 128 * 4);     // scatter accum (reused)
    __bf16* h1   = (__bf16*)take((size_t)N * 128 * 2);    // relu(conv1) bf16
    __bf16* Wt3  = (__bf16*)take((size_t)320 * 320 * 2);  // transposed bf16 weights
    __bf16* Wtc1 = (__bf16*)take((size_t)320 * 128 * 2);
    __bf16* Wtc2 = (__bf16*)take((size_t)128 * 128 * 2);
    float*  pool = (float*)take((size_t)C * 128 * 4);     // sum1+sum2
    float*  cnt  = (float*)take((size_t)C * 4);

    // ---- zero accumulators
    zero_f32_kernel<<<2048, 256, 0, stream>>>(dis, N);
    zero_f32_kernel<<<2048, 256, 0, stream>>>(agg, (long long)N * 128);
    zero_f32_kernel<<<2048, 256, 0, stream>>>(pool, (long long)C * 128);
    zero_f32_kernel<<<64, 256, 0, stream>>>(cnt, C);

    // ---- degree / normalization
    deg_count_kernel<<<(E + 255) / 256, 256, 0, stream>>>(dst, dis, E);
    dis_kernel<<<(N + 255) / 256, 256, 0, stream>>>(dis, N);

    // ---- weight conversion (transpose, f32 -> bf16)
    w_to_bf16_t_kernel<<<(320 * 320 + 255) / 256, 256, 0, stream>>>(W_emb3, Wt3, 320, 320);
    w_to_bf16_t_kernel<<<(320 * 128 + 255) / 256, 256, 0, stream>>>(W_c1, Wtc1, 320, 128);
    w_to_bf16_t_kernel<<<(128 * 128 + 255) / 256, 256, 0, stream>>>(W_c2, Wtc2, 128, 128);

    // ---- embedding frontend
    emb_kernel<<<N, 128, 0, stream>>>(x, id_emb, W_emb1, b_emb1, W_emb2, b_emb2, hcat, N);

    // ---- h = relu(hcat @ W_emb3 + b)  [N,320] -> bf16
    {
        dim3 g((N + 63) / 64, 320 / 32);
        wmma_gemm_kernel<true><<<g, 256, 0, stream>>>(hcat, Wt3, b_emb3, h320, N, 320, 320, 1);
    }

    // ---- conv1: hw = h @ W_conv1  [N,128] f32
    {
        dim3 g((N + 63) / 64, 128 / 32);
        wmma_gemm_kernel<false><<<g, 256, 0, stream>>>(h320, Wtc1, nullptr, hw, N, 320, 128, 0);
    }
    scatter_kernel<<<(E * 32 + 255) / 256, 256, 0, stream>>>(src, dst, dis, hw, agg, E);
    finalize_kernel<<<N, 128, 0, stream>>>(agg, hw, dis, b_c1, comm, pool, cnt, h1, N, 1);

    // ---- conv2: hw = h1 @ W_conv2
    zero_f32_kernel<<<2048, 256, 0, stream>>>(agg, (long long)N * 128);
    {
        dim3 g((N + 63) / 64, 128 / 32);
        wmma_gemm_kernel<false><<<g, 256, 0, stream>>>(h1, Wtc2, nullptr, hw, N, 128, 128, 0);
    }
    scatter_kernel<<<(E * 32 + 255) / 256, 256, 0, stream>>>(src, dst, dis, hw, agg, E);
    finalize_kernel<<<N, 128, 0, stream>>>(agg, hw, dis, b_c2, comm, pool, cnt, nullptr, N, 0);

    // ---- head
    head_kernel<<<C, 128, 0, stream>>>(pool, cnt, W_l1, b_l1, W_l2, b_l2, out, C);
}
